// MultiHeadAttention_33088428048411
// MI455X (gfx1250) — matrix operand order repaired and checked by reading, not compiled, not run on hardware
//
#include <hip/hip_runtime.h>
#include <hip/hip_bf16.h>
#include <stdint.h>

// MHA forward for MI455X (gfx1250, wave32, WMMA).
// B=4, S=2048, E=1024, H=16, DH=64.
#define B_  4
#define S_  2048
#define E_  1024
#define H_  16
#define DH_ 64

typedef __attribute__((ext_vector_type(16))) _Float16 v16h;
typedef __attribute__((ext_vector_type(8)))  float    v8f;

union FragH { v16h v; uint32_t u[8]; };

#if defined(__has_builtin)
#if __has_builtin(__builtin_amdgcn_global_load_async_to_lds_b128)
#define HAVE_ASYNC 1
#endif
#endif
#ifndef HAVE_ASYNC
#define HAVE_ASYNC 0
#endif

#if HAVE_ASYNC
typedef int async_v4i __attribute__((ext_vector_type(4)));
typedef __attribute__((address_space(1))) async_v4i* async_gp;  // global src
typedef __attribute__((address_space(3))) async_v4i* async_lp;  // LDS dst
#endif

__device__ __forceinline__ v8f zero8() {
  v8f z = {0.f,0.f,0.f,0.f,0.f,0.f,0.f,0.f};
  return z;
}

// A-fragment (16x32 f16, M x K), staged row-major [row][32 halves], row stride 40 halves.
// ISA 7.12.2: lane L: row = L%16, half-group h = L/16;
//   VGPR j in 0..3 : K = 8h + 2j, 2j+1    -> dword (4h + j)
//   VGPR j in 4..7 : K = 16 + 8h + 2(j-4) -> dword (8 + 4h + (j-4))
__device__ __forceinline__ v16h lds_load_a(const _Float16* base, int row0) {
  const int lane = threadIdx.x & 31;
  const int r = row0 + (lane & 15);
  const int h = lane >> 4;
  const uint32_t* p = (const uint32_t*)(base + r * 40);
  FragH f;
#pragma unroll
  for (int j = 0; j < 4; ++j) {
    f.u[j]     = p[4*h + j];
    f.u[4 + j] = p[8 + 4*h + j];
  }
  return f.v;
}

// B-fragment (32x16 f16, K x N), staged N-major [n][32 halves of K], stride 40 halves.
// lane L: col n = L%16, h = L/16; same K order as A: VGPR j<4 -> dword (4h + j), j>=4 -> dword (8 + 4h + (j-4))
__device__ __forceinline__ v16h lds_load_b(const _Float16* base, int n0) {
  const int lane = threadIdx.x & 31;
  const int n = n0 + (lane & 15);
  const int h = lane >> 4;
  const uint32_t* p = (const uint32_t*)(base + n * 40);
  FragH f;
#pragma unroll
  for (int j = 0; j < 4; ++j) {
    f.u[j]     = p[4*h + j];
    f.u[4 + j] = p[8 + 4*h + j];
  }
  return f.v;
}

__device__ __forceinline__ v8f wmma_f16(v16h a, v16h b, v8f c) {
  return __builtin_amdgcn_wmma_f32_16x16x32_f16(false, a, false, b, (short)0, c, false, false);
}

// Copy ROWS rows x 32 halves (64B/row) from global (row stride in halves) into
// LDS tile (row stride 40 halves). Uses CDNA5 async global->LDS when available.
template <int ROWS>
__device__ __forceinline__ void stage_tile(const _Float16* __restrict__ g,
                                           int strideHalves,
                                           _Float16* __restrict__ lds, int tid) {
#pragma unroll
  for (int p = 0; p < ROWS / 64; ++p) {  // ROWS*4 chunks of 16B, 256 threads
    int idx = p * 256 + tid;
    int r = idx >> 2;
    int q = (idx & 3) * 8;  // halves offset within the row
#if HAVE_ASYNC
    _Float16* gg = const_cast<_Float16*>(g) + (size_t)r * strideHalves + q;
    __builtin_amdgcn_global_load_async_to_lds_b128(
        (async_gp)gg, (async_lp)(lds + r * 40 + q), 0, 0);
#else
    *(uint4*)(lds + r * 40 + q) =
        *(const uint4*)(g + (size_t)r * strideHalves + q);
#endif
  }
}

__device__ __forceinline__ void async_wait() {
#if HAVE_ASYNC
#if __has_builtin(__builtin_amdgcn_s_wait_asynccnt)
  __builtin_amdgcn_s_wait_asynccnt(0);
#else
  asm volatile("s_wait_asynccnt 0x0" ::: "memory");
#endif
#endif
}

// ---------------------------------------------------------------- converts
__global__ void cvt_f32_to_f16(const float* __restrict__ in,
                               _Float16* __restrict__ out, int n) {
  int i = blockIdx.x * blockDim.x + threadIdx.x;
  int stride = gridDim.x * blockDim.x;
  for (; i < n; i += stride) out[i] = (_Float16)in[i];
}

// ---------------------------------------------------------------- QKV GEMM
// C[h, m, n] = sum_k hid[m,k] * W[h][k,n] + bias[h,n], store f16
// grid: (M/128, H, 3{q,k,v}), block 256 (8 waves), tile 128x64, K-step 32.
__global__ void __launch_bounds__(256) qkv_gemm(
    const _Float16* __restrict__ hid,
    const _Float16* __restrict__ Wq16, const _Float16* __restrict__ Wk16,
    const _Float16* __restrict__ Wv16,
    const float* __restrict__ bq, const float* __restrict__ bk,
    const float* __restrict__ bv,
    _Float16* __restrict__ q16, _Float16* __restrict__ k16,
    _Float16* __restrict__ v16o) {
  __shared__ __align__(16) _Float16 ldsA[128 * 40];
  __shared__ __align__(16) _Float16 ldsB[64 * 40];
  const int mbase = blockIdx.x * 128;
  const int h = blockIdx.y;
  const int sel = blockIdx.z;
  const _Float16* W =
      (sel == 0 ? Wq16 : sel == 1 ? Wk16 : Wv16) + (size_t)h * E_ * DH_;
  const float* bias = (sel == 0 ? bq : sel == 1 ? bk : bv) + h * DH_;
  _Float16* out =
      (sel == 0 ? q16 : sel == 1 ? k16 : v16o) + (size_t)h * B_ * S_ * DH_;

  const int tid = threadIdx.x;
  const int wave = tid >> 5;
  const int mrow0 = (wave >> 1) * 32;  // 0,32,64,96
  const int ncol0 = (wave & 1) * 32;   // 0,32

  v8f acc[2][2];
#pragma unroll
  for (int i = 0; i < 2; ++i)
#pragma unroll
    for (int j = 0; j < 2; ++j) acc[i][j] = zero8();

  for (int k0 = 0; k0 < E_; k0 += 32) {
    stage_tile<128>(hid + (size_t)mbase * E_ + k0, E_, ldsA, tid);
    // stage B transposed: lds[n][kk] = W[k0+kk][n]
#pragma unroll
    for (int p = 0; p < 8; ++p) {
      int idx = p * 256 + tid;
      int kk = idx >> 6;
      int n = idx & 63;
      ldsB[n * 40 + kk] = W[(size_t)(k0 + kk) * DH_ + n];
    }
    async_wait();
    __syncthreads();
    v16h a0 = lds_load_a(ldsA, mrow0);
    v16h a1 = lds_load_a(ldsA, mrow0 + 16);
    v16h b0 = lds_load_b(ldsB, ncol0);
    v16h b1 = lds_load_b(ldsB, ncol0 + 16);
    acc[0][0] = wmma_f16(a0, b0, acc[0][0]);
    acc[0][1] = wmma_f16(a0, b1, acc[0][1]);
    acc[1][0] = wmma_f16(a1, b0, acc[1][0]);
    acc[1][1] = wmma_f16(a1, b1, acc[1][1]);
    __syncthreads();
  }
  const int lane = tid & 31;
  const int cn = lane & 15, ch = lane >> 4;
#pragma unroll
  for (int mi = 0; mi < 2; ++mi)
#pragma unroll
    for (int ni = 0; ni < 2; ++ni) {
      int ncol = ncol0 + ni * 16 + cn;
      float bb = bias[ncol];
#pragma unroll
      for (int j = 0; j < 8; ++j) {
        int row = mbase + mrow0 + mi * 16 + 8 * ch + j;
        out[(size_t)row * DH_ + ncol] = (_Float16)(acc[mi][ni][j] + bb);
      }
    }
}

// ---------------------------------------------------------------- scores
// scores[hb, m, t] = 0.125 * sum_d q[hb,m,d]*k[hb,t,d]
// grid: (S/128, S/128, H*B), block 256, tile 128x128, K=64 in 2 steps.
__global__ void __launch_bounds__(256) scores_gemm(
    const _Float16* __restrict__ q16, const _Float16* __restrict__ k16,
    float* __restrict__ wout) {
  __shared__ __align__(16) _Float16 ldsA[128 * 40];
  __shared__ __align__(16) _Float16 ldsB[128 * 40];
  const int mb = blockIdx.x * 128;
  const int nb = blockIdx.y * 128;
  const int hb = blockIdx.z;
  const _Float16* qb = q16 + (size_t)hb * S_ * DH_;
  const _Float16* kb = k16 + (size_t)hb * S_ * DH_;
  float* ob = wout + (size_t)hb * S_ * S_;

  const int tid = threadIdx.x;
  const int wave = tid >> 5;
  const int mrow0 = (wave >> 1) * 32;  // 0..96
  const int ncol0 = (wave & 1) * 64;   // 0,64

  v8f acc[2][4];
#pragma unroll
  for (int i = 0; i < 2; ++i)
#pragma unroll
    for (int j = 0; j < 4; ++j) acc[i][j] = zero8();

  for (int k0 = 0; k0 < DH_; k0 += 32) {
    stage_tile<128>(qb + (size_t)mb * DH_ + k0, DH_, ldsA, tid);
    stage_tile<128>(kb + (size_t)nb * DH_ + k0, DH_, ldsB, tid);
    async_wait();
    __syncthreads();
    v16h a0 = lds_load_a(ldsA, mrow0);
    v16h a1 = lds_load_a(ldsA, mrow0 + 16);
#pragma unroll
    for (int ni = 0; ni < 4; ++ni) {
      v16h bf = lds_load_b(ldsB, ncol0 + ni * 16);
      acc[0][ni] = wmma_f16(a0, bf, acc[0][ni]);
      acc[1][ni] = wmma_f16(a1, bf, acc[1][ni]);
    }
    __syncthreads();
  }
  const int lane = tid & 31;
  const int cn = lane & 15, ch = lane >> 4;
#pragma unroll
  for (int mi = 0; mi < 2; ++mi)
#pragma unroll
    for (int ni = 0; ni < 4; ++ni)
#pragma unroll
      for (int j = 0; j < 8; ++j) {
        int row = mb + mrow0 + mi * 16 + 8 * ch + j;
        int col = nb + ncol0 + ni * 16 + cn;
        ob[(size_t)row * S_ + col] = acc[mi][ni][j] * 0.125f;
      }
}

// ---------------------------------------------------------------- softmax
// one 256-thread block per (h,b,s) row of 2048, in place, float4 I/O
__global__ void __launch_bounds__(256) softmax_rows(float* __restrict__ w) {
  __shared__ float red[256];
  float* p = w + (size_t)blockIdx.x * S_;
  const int tid = threadIdx.x;
  float4 a0 = ((const float4*)p)[tid];
  float4 a1 = ((const float4*)p)[tid + 256];
  float m = fmaxf(fmaxf(fmaxf(a0.x, a0.y), fmaxf(a0.z, a0.w)),
                  fmaxf(fmaxf(a1.x, a1.y), fmaxf(a1.z, a1.w)));
  red[tid] = m;
  __syncthreads();
  for (int off = 128; off > 0; off >>= 1) {
    if (tid < off) red[tid] = fmaxf(red[tid], red[tid + off]);
    __syncthreads();
  }
  m = red[0];
  __syncthreads();
  a0.x = __expf(a0.x - m); a0.y = __expf(a0.y - m);
  a0.z = __expf(a0.z - m); a0.w = __expf(a0.w - m);
  a1.x = __expf(a1.x - m); a1.y = __expf(a1.y - m);
  a1.z = __expf(a1.z - m); a1.w = __expf(a1.w - m);
  red[tid] = (a0.x + a0.y + a0.z + a0.w) + (a1.x + a1.y + a1.z + a1.w);
  __syncthreads();
  for (int off = 128; off > 0; off >>= 1) {
    if (tid < off) red[tid] += red[tid + off];
    __syncthreads();
  }
  float inv = 1.0f / red[0];
  a0.x *= inv; a0.y *= inv; a0.z *= inv; a0.w *= inv;
  a1.x *= inv; a1.y *= inv; a1.z *= inv; a1.w *= inv;
  ((float4*)p)[tid] = a0;
  ((float4*)p)[tid + 256] = a1;
}

// ---------------------------------------------------------------- attn @ V
// concat16[(b*S+m)*E + h*64 + n] = sum_t weights[hb,m,t]*v[hb,t,n]
// Fused: electrode[b,t] += (1/(H*S)) * sum_{m in tile} weights[hb,m,t]
// grid: (S/128, H*B), block 256, tile 128x64, K=S in steps of 32.
__global__ void __launch_bounds__(256) attn_gemm(
    const float* __restrict__ w, const _Float16* __restrict__ v16i,
    _Float16* __restrict__ concat16, float* __restrict__ eout) {
  __shared__ __align__(16) _Float16 ldsA[128 * 40];
  __shared__ __align__(16) _Float16 ldsB[64 * 40];
  __shared__ float eSum[32];
  const int mb = blockIdx.x * 128;
  const int hb = blockIdx.y;
  const int h = hb >> 2;  // hb = h*B + b, B=4
  const int b = hb & 3;
  const float* wb = w + (size_t)hb * S_ * S_;
  const _Float16* vb = v16i + (size_t)hb * S_ * DH_;

  const int tid = threadIdx.x;
  const int wave = tid >> 5;
  const int mrow0 = (wave >> 1) * 32;
  const int ncol0 = (wave & 1) * 32;
  const int c2 = (tid & 15) * 2;  // this thread's fixed column pair in stage A

  if (tid < 32) eSum[tid] = 0.f;
  __syncthreads();

  v8f acc[2][2];
#pragma unroll
  for (int i = 0; i < 2; ++i)
#pragma unroll
    for (int j = 0; j < 2; ++j) acc[i][j] = zero8();

  for (int k0 = 0; k0 < S_; k0 += 32) {
    // stage A: weights f32 -> packed f16 pairs, plus electrode partials
    float ex = 0.f, ey = 0.f;
#pragma unroll
    for (int p = 0; p < 8; ++p) {
      int idx = p * 256 + tid;
      int r = idx >> 4;  // 8 rows per thread, columns fixed = c2, c2+1
      float2 f = *(const float2*)(wb + (size_t)(mb + r) * S_ + k0 + c2);
      union { _Float16 hh[2]; uint32_t u; } pk;
      pk.hh[0] = (_Float16)f.x;
      pk.hh[1] = (_Float16)f.y;
      *(uint32_t*)(ldsA + r * 40 + c2) = pk.u;
      ex += f.x;
      ey += f.y;
    }
    atomicAdd(&eSum[c2], ex);
    atomicAdd(&eSum[c2 + 1], ey);
    // stage B transposed: lds[n=d][kk=t] = v[k0+kk][n]
#pragma unroll
    for (int p = 0; p < 8; ++p) {
      int idx = p * 256 + tid;
      int kk = idx >> 6;
      int n = idx & 63;
      ldsB[n * 40 + kk] = vb[(size_t)(k0 + kk) * DH_ + n];
    }
    __syncthreads();  // staging + eSum adds complete
    v16h a0 = lds_load_a(ldsA, mrow0);
    v16h a1 = lds_load_a(ldsA, mrow0 + 16);
    v16h b0 = lds_load_b(ldsB, ncol0);
    v16h b1 = lds_load_b(ldsB, ncol0 + 16);
    acc[0][0] = wmma_f16(a0, b0, acc[0][0]);
    acc[0][1] = wmma_f16(a0, b1, acc[0][1]);
    acc[1][0] = wmma_f16(a1, b0, acc[1][0]);
    acc[1][1] = wmma_f16(a1, b1, acc[1][1]);
    // flush electrode partials for this k-step (t = k0 + tid)
    if (tid < 32) {
      float t = eSum[tid];
      eSum[tid] = 0.f;
      atomicAdd(&eout[b * S_ + k0 + tid], t * (1.0f / (H_ * S_)));
    }
    __syncthreads();  // protects LDS tiles and eSum for next iteration
  }
  const int lane = tid & 31;
  const int cn = lane & 15, ch = lane >> 4;
#pragma unroll
  for (int mi = 0; mi < 2; ++mi)
#pragma unroll
    for (int ni = 0; ni < 2; ++ni)
#pragma unroll
      for (int j = 0; j < 8; ++j) {
        int row = mb + mrow0 + mi * 16 + 8 * ch + j;
        int col = h * DH_ + ncol0 + ni * 16 + cn;
        concat16[(size_t)(b * S_ + row) * E_ + col] = (_Float16)acc[mi][ni][j];
      }
}

// ---------------------------------------------------------------- out proj
// x[m, n] = sum_e concat[m,e] * Wo[n,e] + bo[n]
// grid: (B*S/128, E/128), block 256, tile 128x128, K=1024 in steps of 32.
__global__ void __launch_bounds__(256) out_gemm(
    const _Float16* __restrict__ c16, const _Float16* __restrict__ Wo16,
    const float* __restrict__ bo, float* __restrict__ x) {
  __shared__ __align__(16) _Float16 ldsA[128 * 40];
  __shared__ __align__(16) _Float16 ldsB[128 * 40];
  const int mb = blockIdx.x * 128;
  const int nb = blockIdx.y * 128;
  const int tid = threadIdx.x;
  const int wave = tid >> 5;
  const int mrow0 = (wave >> 1) * 32;
  const int ncol0 = (wave & 1) * 64;

  v8f acc[2][4];
#pragma unroll
  for (int i = 0; i < 2; ++i)
#pragma unroll
    for (int j = 0; j < 4; ++j) acc[i][j] = zero8();

  for (int k0 = 0; k0 < E_; k0 += 32) {
    stage_tile<128>(c16 + (size_t)mb * E_ + k0, E_, ldsA, tid);
    stage_tile<128>(Wo16 + (size_t)nb * E_ + k0, E_, ldsB, tid);
    async_wait();
    __syncthreads();
    v16h a0 = lds_load_a(ldsA, mrow0);
    v16h a1 = lds_load_a(ldsA, mrow0 + 16);
#pragma unroll
    for (int ni = 0; ni < 4; ++ni) {
      v16h bf = lds_load_b(ldsB, ncol0 + ni * 16);
      acc[0][ni] = wmma_f16(a0, bf, acc[0][ni]);
      acc[1][ni] = wmma_f16(a1, bf, acc[1][ni]);
    }
    __syncthreads();
  }
  const int lane = tid & 31;
  const int cn = lane & 15, ch = lane >> 4;
#pragma unroll
  for (int mi = 0; mi < 2; ++mi)
#pragma unroll
    for (int ni = 0; ni < 4; ++ni) {
      int col = nb + ncol0 + ni * 16 + cn;
      float bb = bo[col];
#pragma unroll
      for (int j = 0; j < 8; ++j) {
        int row = mb + mrow0 + mi * 16 + 8 * ch + j;
        x[(size_t)row * E_ + col] = acc[mi][ni][j] + bb;
      }
    }
}

// ---------------------------------------------------------------- utility
__global__ void zero_f32(float* __restrict__ p, int n) {
  int i = blockIdx.x * blockDim.x + threadIdx.x;
  if (i < n) p[i] = 0.f;
}

// ---------------------------------------------------------------- launch
extern "C" void kernel_launch(void* const* d_in, const int* in_sizes, int n_in,
                              void* d_out, int out_size, void* d_ws,
                              size_t ws_size, hipStream_t stream) {
  const float* hidden = (const float*)d_in[0];
  const float* Wq = (const float*)d_in[1];
  const float* bq = (const float*)d_in[2];
  const float* Wk = (const float*)d_in[3];
  const float* bk = (const float*)d_in[4];
  const float* Wv = (const float*)d_in[5];
  const float* bv = (const float*)d_in[6];
  const float* Wo = (const float*)d_in[7];
  const float* bo = (const float*)d_in[8];

  float* x = (float*)d_out;                  // [B,S,E]
  float* w = x + (size_t)B_ * S_ * E_;       // [H,B,S,S]
  float* e = w + (size_t)H_ * B_ * S_ * S_;  // [B,S]

  // workspace layout (~92 MB total)
  char* ws = (char*)d_ws;
  _Float16* hidden16 = (_Float16*)ws; ws += (size_t)B_ * S_ * E_ * 2;
  _Float16* Wq16 = (_Float16*)ws;     ws += (size_t)H_ * E_ * DH_ * 2;
  _Float16* Wk16 = (_Float16*)ws;     ws += (size_t)H_ * E_ * DH_ * 2;
  _Float16* Wv16 = (_Float16*)ws;     ws += (size_t)H_ * E_ * DH_ * 2;
  _Float16* Wo16 = (_Float16*)ws;     ws += (size_t)E_ * E_ * 2;
  _Float16* q16 = (_Float16*)ws;      ws += (size_t)H_ * B_ * S_ * DH_ * 2;
  _Float16* k16 = (_Float16*)ws;      ws += (size_t)H_ * B_ * S_ * DH_ * 2;
  _Float16* v16 = (_Float16*)ws;      ws += (size_t)H_ * B_ * S_ * DH_ * 2;
  _Float16* concat16 = (_Float16*)ws; ws += (size_t)B_ * S_ * E_ * 2;

  // 0) f32 -> f16 conversions
  cvt_f32_to_f16<<<2048, 256, 0, stream>>>(hidden, hidden16, B_ * S_ * E_);
  cvt_f32_to_f16<<<1024, 256, 0, stream>>>(Wq, Wq16, H_ * E_ * DH_);
  cvt_f32_to_f16<<<1024, 256, 0, stream>>>(Wk, Wk16, H_ * E_ * DH_);
  cvt_f32_to_f16<<<1024, 256, 0, stream>>>(Wv, Wv16, H_ * E_ * DH_);
  cvt_f32_to_f16<<<1024, 256, 0, stream>>>(Wo, Wo16, E_ * E_);

  // 1) QKV projections (+bias), f16 outputs
  qkv_gemm<<<dim3((B_ * S_) / 128, H_, 3), 256, 0, stream>>>(
      hidden16, Wq16, Wk16, Wv16, bq, bk, bv, q16, k16, v16);

  // 2) scores = scale * q k^T  -> weights slot (raw)
  scores_gemm<<<dim3(S_ / 128, S_ / 128, H_ * B_), 256, 0, stream>>>(q16, k16, w);

  // 3) softmax in place over last dim
  softmax_rows<<<H_ * B_ * S_, 256, 0, stream>>>(w);

  // 4) electrode zero-init, then attn = weights @ v (electrode fused in)
  zero_f32<<<(B_ * S_ + 255) / 256, 256, 0, stream>>>(e, B_ * S_);
  attn_gemm<<<dim3(S_ / 128, H_ * B_), 256, 0, stream>>>(w, v16, concat16, e);

  // 5) x = concat @ Wo^T + bo
  out_gemm<<<dim3((B_ * S_) / 128, E_ / 128), 256, 0, stream>>>(concat16, Wo16, bo, x);
}
